// AttnPool1dWindow_8263517077620
// MI455X (gfx1250) — matrix operand order repaired and checked by reading, not compiled, not run on hardware
//
#include <hip/hip_runtime.h>

#define B_DIM    8
#define C_DIM    512
#define L_DIM    4096
#define LP_DIM   511
#define KSIZE    16
#define STRIDE_P 8
#define NEGV     (-1000000000.0f)
#define ASTR     520           // padded LDS row stride in bf16 elems (1040B -> 4-bank skew/row)
#define AB_ELEMS (4 * 16 * ASTR)       // A strips: 4 waves x 16 rows
#define BBUF_B   (16 * ASTR * 2)       // one B strip buffer, bytes (16,640)
#define BROW_B   (ASTR * 2)            // padded B row, bytes (1040)

typedef __attribute__((ext_vector_type(16))) __bf16 v16bf;
typedef __attribute__((ext_vector_type(8)))  float  v8f;

union Frag {
    struct { uint4 lo, hi; } u;
    v16bf v;
};

static __device__ __forceinline__ unsigned short f2bf(float f) {
    // round-to-nearest-even fp32 -> bf16
    unsigned int u = __float_as_uint(f);
    u += 0x7FFFu + ((u >> 16) & 1u);
    return (unsigned short)(u >> 16);
}

static __device__ __forceinline__ float fast_tanh(float z) {
    // branchless: tanh(|z|) = (1 - e^(-2|z|)) / (1 + e^(-2|z|)); arg <= 0 so never overflows
    const float t = __expf(-2.0f * fabsf(z));
    return copysignf(__fdividef(1.0f - t, 1.0f + t), z);
}

// ---------------------------------------------------------------------------
// Kernel 0: one-time W (512x512 f32) -> bf16 row-major into workspace
// ---------------------------------------------------------------------------
__global__ void __launch_bounds__(256)
wconv_kernel(const float* __restrict__ W, unsigned short* __restrict__ Wbf)
{
    const int i = (blockIdx.x * 256 + threadIdx.x) * 4;   // 65536 threads x 4 = 262144
    float4 f = *(const float4*)(W + i);
    ushort4 h;
    h.x = f2bf(f.x); h.y = f2bf(f.y); h.z = f2bf(f.z); h.w = f2bf(f.w);
    *(ushort4*)(Wbf + i) = h;
}

// ---------------------------------------------------------------------------
// Kernel 1: s[b,l] = v . tanh( X W^T + b ),  X[b*L+l, c] = x[b,c,l]
// M = B*L = 32768 (2048 tiles of 16), N = K = 512.
// 4 waves/block, each wave owns a 16-row M tile. W strips (bf16, pre-converted)
// are DMA'd global->LDS with global_load_async_to_lds_b128, double-buffered so
// the next strip's copy overlaps the current strip's 16 WMMAs.
// ---------------------------------------------------------------------------
__global__ void __launch_bounds__(128)
gemm_tanh_score_kernel(const float* __restrict__ x,
                       const unsigned short* __restrict__ Wbf,
                       const float* __restrict__ bias,
                       const float* __restrict__ vvec,
                       float* __restrict__ s_out)
{
    extern __shared__ unsigned short smem[];
    const int tid  = threadIdx.x;
    const int wave = tid >> 5;
    const int lane = tid & 31;

    unsigned short* Asm = smem + wave * 16 * ASTR;   // this wave's A strip (16 x 512 bf16)
    unsigned short* Bsm = smem + AB_ELEMS;           // B ping-pong buffers (2 x 16 x 512 bf16)
    const unsigned int bsm_lds = (unsigned int)(uintptr_t)(void*)Bsm;  // LDS byte offset
    const unsigned long long wglobal = (unsigned long long)(uintptr_t)Wbf;

    const int mtile = blockIdx.x * 4 + wave;         // 0..2047
    const int m0    = mtile * 16;                    // base bl index
    const int bidx  = m0 >> 12;                      // / L_DIM
    const int l0    = m0 & (L_DIM - 1);

    // ---- stage A: 16 rows of X (x transposed), fp32 -> bf16 into LDS ----
    {
        const int mrow  = lane & 15;
        const int chalf = (lane >> 4) * 4;           // 0 or 4
        const float* xrow = x + ((size_t)bidx * C_DIM) * L_DIM + (size_t)(l0 + mrow);
        unsigned short* arow = Asm + mrow * ASTR;
        for (int c8 = 0; c8 < C_DIM; c8 += 8) {
            const int c = c8 + chalf;
            float f0 = xrow[(size_t)(c + 0) * L_DIM];
            float f1 = xrow[(size_t)(c + 1) * L_DIM];
            float f2 = xrow[(size_t)(c + 2) * L_DIM];
            float f3 = xrow[(size_t)(c + 3) * L_DIM];
            ushort4 h;
            h.x = f2bf(f0); h.y = f2bf(f1); h.z = f2bf(f2); h.w = f2bf(f3);
            *(ushort4*)(arow + c) = h;
        }
    }

    // ---- async DMA of one 16x512 bf16 W strip into LDS buffer buf (0/1) ----
    // thread t copies 8 x 16B chunks: chunk = t + i*128; row = chunk>>6, col16 = chunk&63
    auto issue_b_strip = [&](int nt, int buf) {
        #pragma unroll
        for (int i = 0; i < 8; ++i) {
            const int ch  = tid + i * 128;
            const int row = ch >> 6;
            const int col = ch & 63;
            const unsigned int ldsb =
                bsm_lds + (unsigned int)(buf * BBUF_B + row * BROW_B + col * 16);
            const unsigned int goff =
                (unsigned int)((((nt * 16 + row) << 9) + (col << 3)) * 2);   // bytes
            asm volatile("global_load_async_to_lds_b128 %0, %1, %2"
                         :: "v"(ldsb), "v"(goff), "s"(wglobal) : "memory");
        }
    };

    float sp[8];
#pragma unroll
    for (int r = 0; r < 8; ++r) sp[r] = 0.f;

    const int mloc = lane & 15;
    const int aoff = (lane < 16) ? 0 : 8;    // ISA 16-bit A layout: hi lanes hold K 8-15 / 24-31
    const int boff = (lane < 16) ? 0 : 8;    // ISA 16-bit B layout: hi lanes hold K 8-15 / 24-31
    const unsigned short* arow = Asm + mloc * ASTR;

    // prologue: DMA strip 0 into buffer 0
    issue_b_strip(0, 0);
    asm volatile("s_wait_asynccnt 0x0" ::: "memory");
    __syncthreads();                                  // also orders A-strip ds_stores

    for (int nt = 0; nt < C_DIM / 16; ++nt) {         // 32 N-tiles
        if (nt + 1 < C_DIM / 16)
            issue_b_strip(nt + 1, (nt + 1) & 1);      // overlap DMA with compute

        const unsigned short* brow = Bsm + (nt & 1) * (16 * ASTR) + mloc * ASTR;

        v8f acc0 = {}, acc1 = {};                     // two independent WMMA chains
#pragma unroll 4
        for (int kt = 0; kt < C_DIM / 32; kt += 2) {  // 16 K-steps, 2 per iter
            {
                const int kb = kt * 32;
                Frag af, bf_;
                af.u.lo  = *(const uint4*)(arow + kb + aoff);
                af.u.hi  = *(const uint4*)(arow + kb + aoff + 16);
                bf_.u.lo = *(const uint4*)(brow + kb + boff);
                bf_.u.hi = *(const uint4*)(brow + kb + boff + 16);
                acc0 = __builtin_amdgcn_wmma_f32_16x16x32_bf16(
                    false, af.v, false, bf_.v, (short)0, acc0, false, false);
            }
            {
                const int kb = (kt + 1) * 32;
                Frag af, bf_;
                af.u.lo  = *(const uint4*)(arow + kb + aoff);
                af.u.hi  = *(const uint4*)(arow + kb + aoff + 16);
                bf_.u.lo = *(const uint4*)(brow + kb + boff);
                bf_.u.hi = *(const uint4*)(brow + kb + boff + 16);
                acc1 = __builtin_amdgcn_wmma_f32_16x16x32_bf16(
                    false, af.v, false, bf_.v, (short)0, acc1, false, false);
            }
        }

        // epilogue: h = tanh(acc + b[d]); accumulate h * v[d] into row partials
        const int d = nt * 16 + mloc;                 // column (lane = N)
        const float bn = bias[d];
        const float vn = vvec[d];
#pragma unroll
        for (int r = 0; r < 8; ++r)
            sp[r] += fast_tanh(acc0[r] + acc1[r] + bn) * vn;

        asm volatile("s_wait_asynccnt 0x0" ::: "memory");  // next strip landed
        __syncthreads();                                   // all waves done with this strip
    }

    // reduce over the 16-lane half-wave (sum over N columns); wave32 shuffles
#pragma unroll
    for (int r = 0; r < 8; ++r) {
        float t = sp[r];
        t += __shfl_xor(t, 1, 32);
        t += __shfl_xor(t, 2, 32);
        t += __shfl_xor(t, 4, 32);
        t += __shfl_xor(t, 8, 32);
        sp[r] = t;
    }
    if (lane == 0) {
#pragma unroll
        for (int r = 0; r < 8; ++r) s_out[m0 + r] = sp[r];          // rows 0..7
    } else if (lane == 16) {
#pragma unroll
        for (int r = 0; r < 8; ++r) s_out[m0 + 8 + r] = sp[r];      // rows 8..15
    }
}

// ---------------------------------------------------------------------------
// Kernel 2: per-(b,p) masked windowed softmax of s, then weighted pooling of x
// ---------------------------------------------------------------------------
__global__ void __launch_bounds__(256)
window_pool_kernel(const float* __restrict__ x,
                   const unsigned char* __restrict__ mask,
                   const float* __restrict__ s,
                   float* __restrict__ out)
{
    const int p   = blockIdx.x;
    const int b   = blockIdx.y;
    const int tid = threadIdx.x;
    __shared__ float wsm[KSIZE];

    if (tid < KSIZE) {
        const int l = p * STRIDE_P + tid;
        const bool mk = mask[(size_t)b * L_DIM + l] != 0;
        const float val = mk ? NEGV : s[(size_t)b * L_DIM + l];

        float mx = val;                               // max over the 16-lane group
        mx = fmaxf(mx, __shfl_xor(mx, 1, 32));
        mx = fmaxf(mx, __shfl_xor(mx, 2, 32));
        mx = fmaxf(mx, __shfl_xor(mx, 4, 32));
        mx = fmaxf(mx, __shfl_xor(mx, 8, 32));

        const float e = expf(val - mx);
        float se = e;                                 // softmax denominator (all 16)
        se += __shfl_xor(se, 1, 32);
        se += __shfl_xor(se, 2, 32);
        se += __shfl_xor(se, 4, 32);
        se += __shfl_xor(se, 8, 32);

        float w = e / se;
        if (mk) w = 0.f;                              // zero masked, then renormalize
        float sn = w;
        sn += __shfl_xor(sn, 1, 32);
        sn += __shfl_xor(sn, 2, 32);
        sn += __shfl_xor(sn, 4, 32);
        sn += __shfl_xor(sn, 8, 32);
        sn = fmaxf(sn, 1e-6f);
        wsm[tid] = w / sn;
    }
    __syncthreads();

    float wt[KSIZE];
#pragma unroll
    for (int i = 0; i < KSIZE; ++i) wt[i] = wsm[i];

    for (int c = tid; c < C_DIM; c += 256) {
        const float* xp = x + ((size_t)b * C_DIM + c) * L_DIM + p * STRIDE_P;
        float4 a0 = *(const float4*)(xp + 0);
        float4 a1 = *(const float4*)(xp + 4);
        float4 a2 = *(const float4*)(xp + 8);
        float4 a3 = *(const float4*)(xp + 12);
        float acc = a0.x * wt[0]  + a0.y * wt[1]  + a0.z * wt[2]  + a0.w * wt[3]
                  + a1.x * wt[4]  + a1.y * wt[5]  + a1.z * wt[6]  + a1.w * wt[7]
                  + a2.x * wt[8]  + a2.y * wt[9]  + a2.z * wt[10] + a2.w * wt[11]
                  + a3.x * wt[12] + a3.y * wt[13] + a3.z * wt[14] + a3.w * wt[15];
        out[((size_t)b * C_DIM + c) * LP_DIM + p] = acc;
    }
}

// ---------------------------------------------------------------------------
extern "C" void kernel_launch(void* const* d_in, const int* in_sizes, int n_in,
                              void* d_out, int out_size, void* d_ws, size_t ws_size,
                              hipStream_t stream) {
    const float*         x    = (const float*)d_in[0];
    const unsigned char* mask = (const unsigned char*)d_in[1];  // jax bool -> 1 byte
    const float*         Wm   = (const float*)d_in[2];
    const float*         bias = (const float*)d_in[3];
    const float*         vvec = (const float*)d_in[4];

    // workspace layout: s scores (B*L f32 = 128KB) | W bf16 (512x512 = 512KB)
    float*          s   = (float*)d_ws;
    unsigned short* Wbf = (unsigned short*)((char*)d_ws + (size_t)B_DIM * L_DIM * sizeof(float));
    float*          out = (float*)d_out;

    const size_t lds_bytes =
        (size_t)AB_ELEMS * sizeof(unsigned short) + 2 * (size_t)BBUF_B;  // 99,840 B

    wconv_kernel<<<dim3(256), dim3(256), 0, stream>>>(Wm, Wbf);
    gemm_tanh_score_kernel<<<dim3(512), dim3(128), lds_bytes, stream>>>(x, Wbf, bias, vvec, s);
    window_pool_kernel<<<dim3(LP_DIM, B_DIM), dim3(256), 0, stream>>>(x, mask, s, out);
}
